// attentive_MotifPool_42322607734793
// MI455X (gfx1250) — matrix operand order repaired and checked by reading, not compiled, not run on hardware
//
#include <hip/hip_runtime.h>
#include <hip/hip_bf16.h>
#include <math.h>

#define N_NODES 20000
#define EDGES   320000
#define HID     256
#define OUTC    128
#define NEGSLOPE 0.01f

typedef __attribute__((ext_vector_type(16))) __bf16 v16bf;
typedef __attribute__((ext_vector_type(8)))  __bf16 v8bf;
typedef __attribute__((ext_vector_type(8)))  float  v8f;

#define WMMA_BF16(a, b, c) \
  __builtin_amdgcn_wmma_f32_16x16x32_bf16(false, (a), false, (b), (short)0, (c), false, false)

union ABFrag { v16bf v; v8bf h[2]; };

// A-matrix (16x32 bf16, row-major source, leading dim ld):
// lanes 0-15: row=lane, K chunks {0..7} and {16..23}
// lanes 16-31: row=lane-16, K chunks {8..15} and {24..31}
__device__ inline v16bf load_a_frag(const __bf16* tile, int ld, int lane) {
  const int row   = lane & 15;
  const int kbase = (lane & 16) ? 8 : 0;
  ABFrag u;
  u.h[0] = *(const v8bf*)(tile + (size_t)row * ld + kbase);
  u.h[1] = *(const v8bf*)(tile + (size_t)row * ld + kbase + 16);
  return u.v;
}

// B-matrix (32x16 bf16) built from row-major W [Mcols, K]; B[k][n] = W[n][k].
// lanes 0-15: col n=lane holds W[n][k0..k0+7],W[n][k0+16..k0+23]; lanes 16-31: W[n][k0+8..k0+15],W[n][k0+24..k0+31]
__device__ inline v16bf load_b_frag(const __bf16* tile, int ld, int lane) {
  const int col   = lane & 15;
  const int kbase = (lane & 16) ? 8 : 0;
  ABFrag u;
  u.h[0] = *(const v8bf*)(tile + (size_t)col * ld + kbase);
  u.h[1] = *(const v8bf*)(tile + (size_t)col * ld + kbase + 16);
  return u.v;
}

// ---------------- generic GEMM: C[rows, Mcols] = A[rows,K](bf16) * W[Mcols,K]^T + bias ---
template <int TPW>
__global__ void gemm_rowmajor_bt(const __bf16* __restrict__ A,
                                 const __bf16* __restrict__ W,
                                 const float*  __restrict__ bias,  // nullable
                                 float* __restrict__ C, int K, int Mcols) {
  const int lane = threadIdx.x & 31;
  const int wave = threadIdx.x >> 5;
  const int m0   = blockIdx.x * 16;
  const int ct0  = (blockIdx.y * 8 + wave) * TPW;

  v8f acc[TPW] = {};
  for (int k0 = 0; k0 < K; k0 += 32) {
    v16bf a = load_a_frag(A + (size_t)m0 * K + k0, K, lane);
#pragma unroll
    for (int t = 0; t < TPW; ++t) {
      v16bf b = load_b_frag(W + (size_t)(ct0 + t) * 16 * K + k0, K, lane);
      acc[t]  = WMMA_BF16(a, b, acc[t]);
    }
  }
  const int col   = lane & 15;
  const int rbase = m0 + ((lane & 16) ? 8 : 0);
#pragma unroll
  for (int t = 0; t < TPW; ++t) {
    const int n  = (ct0 + t) * 16 + col;
    const float bv = bias ? bias[n] : 0.0f;
#pragma unroll
    for (int j = 0; j < 8; ++j)
      C[(size_t)(rbase + j) * Mcols + n] = acc[t][j] + bv;
  }
}

// ---------------- fused GRU: 6 WMMA accumulators per tile, elementwise in-register ------
__global__ void gru_fused(const __bf16* __restrict__ Mbf, const __bf16* __restrict__ Hbf,
                          const float* __restrict__ Hold,
                          const __bf16* __restrict__ Wih, const __bf16* __restrict__ Whh,
                          const float* __restrict__ bih, const float* __restrict__ bhh,
                          float* __restrict__ Hnew, __bf16* __restrict__ HbfNew) {
  const int lane = threadIdx.x & 31;
  const int wave = threadIdx.x >> 5;
  const int m0   = blockIdx.x * 16;
  const int n0   = (blockIdx.y * 8 + wave) * 16;  // 16 tiles across H=256

  v8f ax[3] = {};  // r, z, n gates from W_ih * m
  v8f ah[3] = {};  // r, z, n gates from W_hh * h
  for (int k0 = 0; k0 < HID; k0 += 32) {
    v16bf am = load_a_frag(Mbf + (size_t)m0 * HID + k0, HID, lane);
    v16bf av = load_a_frag(Hbf + (size_t)m0 * HID + k0, HID, lane);
#pragma unroll
    for (int g = 0; g < 3; ++g) {
      v16bf b1 = load_b_frag(Wih + (size_t)(g * HID + n0) * HID + k0, HID, lane);
      ax[g]    = WMMA_BF16(am, b1, ax[g]);
      v16bf b2 = load_b_frag(Whh + (size_t)(g * HID + n0) * HID + k0, HID, lane);
      ah[g]    = WMMA_BF16(av, b2, ah[g]);
    }
  }
  const int col   = n0 + (lane & 15);
  const int rbase = m0 + ((lane & 16) ? 8 : 0);
  const float br0 = bih[col], br1 = bih[HID + col], br2 = bih[2 * HID + col];
  const float bh0 = bhh[col], bh1 = bhh[HID + col], bh2 = bhh[2 * HID + col];
#pragma unroll
  for (int j = 0; j < 8; ++j) {
    const int row = rbase + j;
    const float xr = ax[0][j] + br0, xz = ax[1][j] + br1, xn = ax[2][j] + br2;
    const float hr = ah[0][j] + bh0, hz = ah[1][j] + bh1, hn = ah[2][j] + bh2;
    const float r  = 1.0f / (1.0f + expf(-(xr + hr)));
    const float z  = 1.0f / (1.0f + expf(-(xz + hz)));
    const float nn = tanhf(xn + r * hn);
    const float hp = Hold[(size_t)row * HID + col];
    const float hv = (1.0f - z) * nn + z * hp;
    Hnew[(size_t)row * HID + col]   = hv;
    HbfNew[(size_t)row * HID + col] = (__bf16)hv;
  }
}

// ---------------- per-node alpha dots + buffer init ------------------------------------
__global__ void node_alpha(const float* __restrict__ xp, const float* __restrict__ att_s,
                           const float* __restrict__ att_d,
                           float* __restrict__ as_, float* __restrict__ ad_,
                           int* __restrict__ amaxI, float* __restrict__ asum,
                           float* __restrict__ msg) {
  const int i = blockIdx.x;
  const int t = threadIdx.x;  // 0..255
  const float v = xp[(size_t)i * HID + t];
  msg[(size_t)i * HID + t] = 0.0f;
  float ps = v * att_s[t];
  float pd = v * att_d[t];
  for (int off = 16; off; off >>= 1) {
    ps += __shfl_down(ps, off, 32);
    pd += __shfl_down(pd, off, 32);
  }
  __shared__ float ss[8], sd[8];
  const int wave = t >> 5, lane = t & 31;
  if (lane == 0) { ss[wave] = ps; sd[wave] = pd; }
  __syncthreads();
  if (t == 0) {
    float S = 0.0f, D = 0.0f;
    for (int w = 0; w < 8; ++w) { S += ss[w]; D += sd[w]; }
    as_[i]   = S;
    ad_[i]   = D;
    amaxI[i] = (int)0x80000000;  // ordered-int -inf
    asum[i]  = 0.0f;
  }
}

// ordered-int mapping for float atomicMax (involution)
__device__ inline int   f2ord(float f) { int i = __float_as_int(f); return i ^ ((i >> 31) & 0x7fffffff); }
__device__ inline float ord2f(int o)   { return __int_as_float(o ^ ((o >> 31) & 0x7fffffff)); }

__global__ void edge_logits(const int* __restrict__ src, const int* __restrict__ dst,
                            const float* __restrict__ as_, const float* __restrict__ ad_,
                            float* __restrict__ ae, int* __restrict__ amaxI) {
  const int e = blockIdx.x * blockDim.x + threadIdx.x;
  if (e >= EDGES) return;
  const int s = src[e], d = dst[e];
  float a = as_[s] + ad_[d];
  a = (a > 0.0f) ? a : NEGSLOPE * a;
  ae[e] = a;
  atomicMax(&amaxI[d], f2ord(a));
}

__global__ void edge_exp(const int* __restrict__ dst, float* __restrict__ ae,
                         const int* __restrict__ amaxI, float* __restrict__ asum) {
  const int e = blockIdx.x * blockDim.x + threadIdx.x;
  if (e >= EDGES) return;
  const int d = dst[e];
  const float ex = expf(ae[e] - ord2f(amaxI[d]));
  ae[e] = ex;
  atomicAdd(&asum[d], ex);
}

// 4 edges per 256-thread block; 64 lanes x float4 per edge (coalesced row reads)
__global__ void edge_scatter(const int* __restrict__ src, const int* __restrict__ dst,
                             const float* __restrict__ ae, const float* __restrict__ asum,
                             const float* __restrict__ xp, float* __restrict__ msg) {
  const int e = blockIdx.x * 4 + (threadIdx.x >> 6);
  if (e >= EDGES) return;
  const int c = (threadIdx.x & 63) * 4;
  const int s = src[e], d = dst[e];
  const float coef = ae[e] / fmaxf(asum[d], 1e-16f);
  const float4 xv  = *(const float4*)(xp + (size_t)s * HID + c);
  float* out = msg + (size_t)d * HID + c;
  atomicAdd(out + 0, coef * xv.x);
  atomicAdd(out + 1, coef * xv.y);
  atomicAdd(out + 2, coef * xv.z);
  atomicAdd(out + 3, coef * xv.w);
}

__global__ void msg_elu_cvt(const float* __restrict__ msg, const float* __restrict__ gbias,
                            __bf16* __restrict__ mbf) {
  const size_t idx = (size_t)blockIdx.x * blockDim.x + threadIdx.x;
  if (idx >= (size_t)N_NODES * HID) return;
  float v = msg[idx] + gbias[idx & (HID - 1)];
  v = (v > 0.0f) ? v : expm1f(v);  // ELU(alpha=1)
  mbf[idx] = (__bf16)v;
}

__global__ void f32_to_bf16(const float* __restrict__ src, __bf16* __restrict__ dst, int n) {
  const int i = blockIdx.x * blockDim.x + threadIdx.x;
  if (i < n) dst[i] = (__bf16)src[i];
}

// =======================================================================================
extern "C" void kernel_launch(void* const* d_in, const int* in_sizes, int n_in,
                              void* d_out, int out_size, void* d_ws, size_t ws_size,
                              hipStream_t stream) {
  const float* x_clique = (const float*)d_in[0];
  const float* W_gat    = (const float*)d_in[1];
  const float* att_src  = (const float*)d_in[2];
  const float* att_dst  = (const float*)d_in[3];
  const float* gat_bias = (const float*)d_in[4];
  const float* W_ih     = (const float*)d_in[5];
  const float* W_hh     = (const float*)d_in[6];
  const float* b_ih     = (const float*)d_in[7];
  const float* b_hh     = (const float*)d_in[8];
  const float* lin_W    = (const float*)d_in[9];
  const float* lin_b    = (const float*)d_in[10];
  const int*   eidx     = (const int*)d_in[14];   // clique_edge_index (2,E)
  const int*   e_src    = eidx;
  const int*   e_dst    = eidx + EDGES;

  // ---- workspace layout (256B-aligned) ----
  char*  base = (char*)d_ws;
  size_t off  = 0;
  auto alloc = [&](size_t bytes) -> void* {
    void* p = base + off;
    off += (bytes + 255) & ~(size_t)255;
    return p;
  };
  __bf16* Wg_bf  = (__bf16*)alloc((size_t)HID * HID * 2);
  __bf16* Wih_bf = (__bf16*)alloc((size_t)3 * HID * HID * 2);
  __bf16* Whh_bf = (__bf16*)alloc((size_t)3 * HID * HID * 2);
  __bf16* lin_bf = (__bf16*)alloc((size_t)OUTC * HID * 2);
  __bf16* hbfA   = (__bf16*)alloc((size_t)N_NODES * HID * 2);
  __bf16* hbfB   = (__bf16*)alloc((size_t)N_NODES * HID * 2);
  __bf16* m_bf   = (__bf16*)alloc((size_t)N_NODES * HID * 2);
  float*  xp     = (float*) alloc((size_t)N_NODES * HID * 4);
  float*  msg    = (float*) alloc((size_t)N_NODES * HID * 4);
  float*  hA     = (float*) alloc((size_t)N_NODES * HID * 4);
  float*  hB     = (float*) alloc((size_t)N_NODES * HID * 4);
  float*  as_    = (float*) alloc((size_t)N_NODES * 4);
  float*  ad_    = (float*) alloc((size_t)N_NODES * 4);
  float*  asum   = (float*) alloc((size_t)N_NODES * 4);
  int*    amaxI  = (int*)   alloc((size_t)N_NODES * 4);
  float*  ae     = (float*) alloc((size_t)EDGES * 4);
  (void)ws_size; (void)in_sizes; (void)n_in; (void)out_size;

  const int rowTiles = N_NODES / 16;  // 1250 (exact)

  // ---- weight / input conversion to bf16 ----
  f32_to_bf16<<<(HID * HID + 255) / 256, 256, 0, stream>>>(W_gat, Wg_bf, HID * HID);
  f32_to_bf16<<<(3 * HID * HID + 255) / 256, 256, 0, stream>>>(W_ih, Wih_bf, 3 * HID * HID);
  f32_to_bf16<<<(3 * HID * HID + 255) / 256, 256, 0, stream>>>(W_hh, Whh_bf, 3 * HID * HID);
  f32_to_bf16<<<(OUTC * HID + 255) / 256, 256, 0, stream>>>(lin_W, lin_bf, OUTC * HID);
  f32_to_bf16<<<(N_NODES * HID + 255) / 256, 256, 0, stream>>>(x_clique, hbfA, N_NODES * HID);

  const float* hold = x_clique;
  const __bf16* hbf = hbfA;
  for (int t = 0; t < 3; ++t) {
    // xp = h @ W_gat^T (WMMA)
    gemm_rowmajor_bt<2><<<dim3(rowTiles, 1), 256, 0, stream>>>(hbf, Wg_bf, nullptr, xp, HID, HID);
    // per-node alpha dots, init amax/asum/msg
    node_alpha<<<N_NODES, 256, 0, stream>>>(xp, att_src, att_dst, as_, ad_, amaxI, asum, msg);
    // segment softmax + weighted scatter
    edge_logits<<<(EDGES + 255) / 256, 256, 0, stream>>>(e_src, e_dst, as_, ad_, ae, amaxI);
    edge_exp<<<(EDGES + 255) / 256, 256, 0, stream>>>(e_dst, ae, amaxI, asum);
    edge_scatter<<<EDGES / 4, 256, 0, stream>>>(e_src, e_dst, ae, asum, xp, msg);
    // bias + ELU + cvt to bf16
    msg_elu_cvt<<<(N_NODES * HID + 255) / 256, 256, 0, stream>>>(msg, gat_bias, m_bf);
    // fused GRU (6 WMMA streams), ping-pong h buffers
    float*  hnew   = (t & 1) ? hB : hA;
    __bf16* hbfnew = (hbf == hbfA) ? hbfB : hbfA;
    gru_fused<<<dim3(rowTiles, 2), 256, 0, stream>>>(m_bf, hbf, hold, Wih_bf, Whh_bf,
                                                     b_ih, b_hh, hnew, hbfnew);
    hold = hnew;
    hbf  = hbfnew;
  }
  // final projection: out = h @ lin_W^T + lin_b
  gemm_rowmajor_bt<1><<<dim3(rowTiles, 1), 256, 0, stream>>>(hbf, lin_bf, lin_b,
                                                             (float*)d_out, HID, OUTC);
}
